// SequenceGRU_86114094285189
// MI455X (gfx1250) — matrix operand order repaired and checked by reading, not compiled, not run on hardware
//
#include <hip/hip_runtime.h>
#include <hip/hip_bf16.h>
#include <stdint.h>
#include <stddef.h>

// ---------------------------------------------------------------------------
// 3-layer LSTM (B=64, S=2048, IN=300, H=1024, OUT=128) for gfx1250 (MI455X).
// bf16 WMMA GEMMs (f32 accumulate), persistent kernel + grid barrier per step.
// ---------------------------------------------------------------------------

typedef __bf16 bf16_t;
typedef bf16_t v16bf __attribute__((ext_vector_type(16)));
typedef bf16_t v8bf  __attribute__((ext_vector_type(8)));
typedef float  v8f   __attribute__((ext_vector_type(8)));

#define B_    64
#define S_    2048
#define IN_   300
#define INP_  320          // padded input K (multiple of 32)
#define H_    1024
#define G4_   4096         // 4*H
#define OUT_  128
#define NWG   32           // persistent grid: 32 WGs x 256 thr = 256 waves
#define NTHR  256

// ---------------------------------------------------------------------------
// math helpers
// ---------------------------------------------------------------------------
__device__ __forceinline__ float sigmoidf_(float x) {
  return 1.0f / (1.0f + __expf(-x));
}
__device__ __forceinline__ float tanhf_(float x) {
  x = fminf(15.0f, fmaxf(-15.0f, x));
  float e = __expf(2.0f * x);
  return (e - 1.0f) / (e + 1.0f);
}

// ---------------------------------------------------------------------------
// WMMA fragment loaders (per CDNA5 ISA 7.12.2 VGPR layouts, wave32)
//   A 16x32 bf16: lane L holds row M=L&15, K = (L>>4)*8 + {0..7, 16..23}
//   B 32x16 bf16: lane L holds col N=L&15, K = (L>>4)*8 + {0..7, 16..23}
// Both are contiguous along K for row-major (rows x K) storage.
// ---------------------------------------------------------------------------
__device__ __forceinline__ v16bf load_afrag(const bf16_t* base, int ld,
                                            int r0, int k0, int lane) {
  int m  = lane & 15;
  int kb = (lane >> 4) * 8;
  const bf16_t* p = base + (size_t)(r0 + m) * ld + k0 + kb;
  v8bf lo = *(const v8bf*)p;
  v8bf hi = *(const v8bf*)(p + 16);
  return __builtin_shufflevector(lo, hi, 0,1,2,3,4,5,6,7,8,9,10,11,12,13,14,15);
}

__device__ __forceinline__ v16bf load_bfrag(const bf16_t* w, int ld,
                                            int n0, int k0, int lane) {
  int n  = lane & 15;
  int kb = (lane >> 4) * 8;
  const bf16_t* p = w + (size_t)(n0 + n) * ld + k0 + kb;
  v8bf lo = *(const v8bf*)p;
  v8bf hi = *(const v8bf*)(p + 16);
  return __builtin_shufflevector(lo, hi, 0,1,2,3,4,5,6,7,8,9,10,11,12,13,14,15);
}

__device__ __forceinline__ v8f wmma_bf16(v16bf a, v16bf b, v8f c) {
  return __builtin_amdgcn_wmma_f32_16x16x32_bf16(false, a, false, b,
                                                 (short)0, c, false, false);
}

// ---------------------------------------------------------------------------
// grid-wide barrier (atomic sense-reversal) -- all NWG blocks resident
// bar[0] = arrive count, bar[1] = generation
// ---------------------------------------------------------------------------
__device__ __forceinline__ void grid_sync(unsigned* bar) {
  __syncthreads();
  if (threadIdx.x == 0) {
    __threadfence();
    unsigned g = __hip_atomic_load(&bar[1], __ATOMIC_ACQUIRE,
                                   __HIP_MEMORY_SCOPE_AGENT);
    unsigned a = __hip_atomic_fetch_add(&bar[0], 1u, __ATOMIC_ACQ_REL,
                                        __HIP_MEMORY_SCOPE_AGENT);
    if (a == gridDim.x - 1) {
      __hip_atomic_store(&bar[0], 0u, __ATOMIC_RELAXED,
                         __HIP_MEMORY_SCOPE_AGENT);
      __hip_atomic_fetch_add(&bar[1], 1u, __ATOMIC_RELEASE,
                             __HIP_MEMORY_SCOPE_AGENT);
    } else {
      while (__hip_atomic_load(&bar[1], __ATOMIC_ACQUIRE,
                               __HIP_MEMORY_SCOPE_AGENT) == g) {
        __builtin_amdgcn_s_sleep(2);
      }
    }
    __threadfence();
  }
  __syncthreads();
}

// ---------------------------------------------------------------------------
// One LSTM layer step for this wave's quad tile:
//   gates[R0:R0+16, J0 + g*H] = Xin @ Wih^T + Hprev @ Whh^T + bias
// Each wave owns all 4 gates for its 16x16 (batch x hidden) slice, so the
// nonlinearity + c/h update happen entirely in-register.
// ---------------------------------------------------------------------------
__device__ __forceinline__ void lstm_layer(const bf16_t* __restrict__ Xin, int Kin,
                                           const bf16_t* __restrict__ Wih,
                                           const bf16_t* __restrict__ Hprev,
                                           const bf16_t* __restrict__ Whh,
                                           const float*  __restrict__ bias,
                                           float*  __restrict__ c,
                                           bf16_t* __restrict__ Hnew,
                                           int R0, int J0, int lane) {
  v8f acc[4] = {};

  // input-to-hidden contribution
  for (int kc = 0; kc < Kin; kc += 32) {
    v16bf a = load_afrag(Xin, Kin, R0, kc, lane);
#pragma unroll
    for (int g = 0; g < 4; ++g) {
      v16bf b = load_bfrag(Wih, Kin, g * H_ + J0, kc, lane);
      acc[g] = wmma_bf16(a, b, acc[g]);
    }
  }
  // hidden-to-hidden contribution
  for (int kc = 0; kc < H_; kc += 32) {
    v16bf a = load_afrag(Hprev, H_, R0, kc, lane);
#pragma unroll
    for (int g = 0; g < 4; ++g) {
      v16bf b = load_bfrag(Whh, H_, g * H_ + J0, kc, lane);
      acc[g] = wmma_bf16(a, b, acc[g]);
    }
  }

  int n  = lane & 15;
  int mh = (lane >> 4) * 8;
  float bi = bias[0 * H_ + J0 + n];
  float bf = bias[1 * H_ + J0 + n];
  float bg = bias[2 * H_ + J0 + n];
  float bo = bias[3 * H_ + J0 + n];

#pragma unroll
  for (int r = 0; r < 8; ++r) {
    int m = R0 + mh + r;          // batch row
    int j = J0 + n;               // hidden unit
    float iv = sigmoidf_(acc[0][r] + bi);
    float fv = sigmoidf_(acc[1][r] + bf);
    float gv = tanhf_  (acc[2][r] + bg);
    float ov = sigmoidf_(acc[3][r] + bo);
    float cold = c[m * H_ + j];
    float cnew = fv * cold + iv * gv;
    c[m * H_ + j] = cnew;
    Hnew[m * H_ + j] = (bf16_t)(ov * tanhf_(cnew));
  }
}

// ---------------------------------------------------------------------------
// Kernel parameter block
// ---------------------------------------------------------------------------
struct KParams {
  const bf16_t* xp;                 // [S][B][INP_] padded bf16 input
  const bf16_t* Wih1; const bf16_t* Whh1;
  const bf16_t* Wih2; const bf16_t* Whh2;
  const bf16_t* Wih3; const bf16_t* Whh3;
  const bf16_t* Wo;
  const float* b1; const float* b2; const float* b3; const float* bo;
  bf16_t* h1[2]; bf16_t* h2[2]; bf16_t* h3[2];   // double-buffered states
  float* c1; float* c2; float* c3;
  float* out;
  unsigned* bar;
};

__global__ void __launch_bounds__(NTHR)
lstm_persistent(KParams P) {
  int tid   = blockIdx.x * blockDim.x + threadIdx.x;
  int gwave = tid >> 5;                 // 0..255
  int lane  = threadIdx.x & 31;
  int R0 = (gwave >> 6) * 16;           // batch tile row (0,16,32,48)
  int J0 = (gwave & 63) * 16;           // hidden tile col (0..1008)

  for (int t = 1; t < S_; ++t) {
    const bf16_t* xt = P.xp + (size_t)t * B_ * INP_;
    int rd = (t - 1) & 1, wr = t & 1;

    lstm_layer(xt,       INP_, P.Wih1, P.h1[rd], P.Whh1, P.b1, P.c1, P.h1[wr], R0, J0, lane);
    grid_sync(P.bar);
    lstm_layer(P.h1[wr], H_,   P.Wih2, P.h2[rd], P.Whh2, P.b2, P.c2, P.h2[wr], R0, J0, lane);
    grid_sync(P.bar);
    lstm_layer(P.h2[wr], H_,   P.Wih3, P.h3[rd], P.Whh3, P.b3, P.c3, P.h3[wr], R0, J0, lane);
    grid_sync(P.bar);

    // output projection: out[:, t, :] = h3 @ Wo^T + bo  (64x128, 32 tiles)
    if (gwave < 32) {
      int rt = gwave >> 3;              // 0..3 (batch tile)
      int ct = gwave & 7;               // 0..7 (out tile)
      v8f acc = {};
      for (int kc = 0; kc < H_; kc += 32) {
        v16bf a = load_afrag(P.h3[wr], H_, rt * 16, kc, lane);
        v16bf b = load_bfrag(P.Wo,     H_, ct * 16, kc, lane);
        acc = wmma_bf16(a, b, acc);
      }
      int n   = ct * 16 + (lane & 15);
      float bb = P.bo[n];
      int mh  = (lane >> 4) * 8;
#pragma unroll
      for (int r = 0; r < 8; ++r) {
        int b_idx = rt * 16 + mh + r;
        P.out[(size_t)b_idx * S_ * OUT_ + (size_t)t * OUT_ + n] = acc[r] + bb;
      }
    }
    grid_sync(P.bar);
  }
}

// ---------------------------------------------------------------------------
// Prep kernels (re-run every launch; deterministic under graph replay)
// ---------------------------------------------------------------------------
__global__ void k_convert_pad(const float* __restrict__ src, bf16_t* __restrict__ dst,
                              int rows, int csrc, int cdst) {
  size_t n = (size_t)rows * cdst;
  for (size_t i = blockIdx.x * (size_t)blockDim.x + threadIdx.x; i < n;
       i += (size_t)gridDim.x * blockDim.x) {
    int r = (int)(i / cdst), k = (int)(i % cdst);
    dst[i] = (k < csrc) ? (bf16_t)src[(size_t)r * csrc + k] : (bf16_t)0.0f;
  }
}

__global__ void k_convert_x(const float* __restrict__ x, bf16_t* __restrict__ xp) {
  size_t n = (size_t)S_ * B_ * INP_;
  for (size_t i = blockIdx.x * (size_t)blockDim.x + threadIdx.x; i < n;
       i += (size_t)gridDim.x * blockDim.x) {
    int k = (int)(i % INP_);
    size_t tb = i / INP_;
    int b = (int)(tb % B_);
    int t = (int)(tb / B_);
    xp[i] = (k < IN_) ? (bf16_t)x[(size_t)b * S_ * IN_ + (size_t)t * IN_ + k]
                      : (bf16_t)0.0f;
  }
}

__global__ void k_bias_add(const float* a, const float* b, float* dst, int n) {
  for (int i = blockIdx.x * blockDim.x + threadIdx.x; i < n;
       i += gridDim.x * blockDim.x)
    dst[i] = a[i] + (b ? b[i] : 0.0f);
}

__global__ void k_zero_f32(float* p, size_t n) {
  for (size_t i = blockIdx.x * (size_t)blockDim.x + threadIdx.x; i < n;
       i += (size_t)gridDim.x * blockDim.x)
    p[i] = 0.0f;
}

__global__ void k_zero_bf16(bf16_t* p, size_t n) {
  for (size_t i = blockIdx.x * (size_t)blockDim.x + threadIdx.x; i < n;
       i += (size_t)gridDim.x * blockDim.x)
    p[i] = (bf16_t)0.0f;
}

__global__ void k_out_t0(float* out) {
  int i = blockIdx.x * blockDim.x + threadIdx.x;
  if (i < B_ * OUT_) {
    int b = i / OUT_, n = i % OUT_;
    out[(size_t)b * S_ * OUT_ + n] = 0.0f;  // out[:, 0, :] = 0
  }
}

__global__ void k_zero_u32(unsigned* p, int n) {
  int i = blockIdx.x * blockDim.x + threadIdx.x;
  if (i < n) p[i] = 0u;
}

// ---------------------------------------------------------------------------
// host launcher
// ---------------------------------------------------------------------------
extern "C" void kernel_launch(void* const* d_in, const int* in_sizes, int n_in,
                              void* d_out, int out_size, void* d_ws, size_t ws_size,
                              hipStream_t stream) {
  const float* x    = (const float*)d_in[0];
  const float* Wih1 = (const float*)d_in[1];
  const float* Whh1 = (const float*)d_in[2];
  const float* bih1 = (const float*)d_in[3];
  const float* bhh1 = (const float*)d_in[4];
  const float* Wih2 = (const float*)d_in[5];
  const float* Whh2 = (const float*)d_in[6];
  const float* bih2 = (const float*)d_in[7];
  const float* bhh2 = (const float*)d_in[8];
  const float* Wih3 = (const float*)d_in[9];
  const float* Whh3 = (const float*)d_in[10];
  const float* bih3 = (const float*)d_in[11];
  const float* bhh3 = (const float*)d_in[12];
  const float* Wo   = (const float*)d_in[13];
  const float* bo   = (const float*)d_in[14];
  float* out = (float*)d_out;

  char* ws = (char*)d_ws;
  size_t off = 0;
  auto alloc = [&](size_t bytes) -> char* {
    char* p = ws + off;
    off += (bytes + 255) & ~(size_t)255;
    return p;
  };

  bf16_t* xp    = (bf16_t*)alloc((size_t)S_ * B_ * INP_ * 2);
  bf16_t* Wih1b = (bf16_t*)alloc((size_t)G4_ * INP_ * 2);
  bf16_t* Whh1b = (bf16_t*)alloc((size_t)G4_ * H_ * 2);
  bf16_t* Wih2b = (bf16_t*)alloc((size_t)G4_ * H_ * 2);
  bf16_t* Whh2b = (bf16_t*)alloc((size_t)G4_ * H_ * 2);
  bf16_t* Wih3b = (bf16_t*)alloc((size_t)G4_ * H_ * 2);
  bf16_t* Whh3b = (bf16_t*)alloc((size_t)G4_ * H_ * 2);
  bf16_t* Wob   = (bf16_t*)alloc((size_t)OUT_ * H_ * 2);
  float*  b1    = (float*)alloc(G4_ * 4);
  float*  b2    = (float*)alloc(G4_ * 4);
  float*  b3    = (float*)alloc(G4_ * 4);
  float*  bob   = (float*)alloc(OUT_ * 4);
  bf16_t* h1[2] = { (bf16_t*)alloc((size_t)B_ * H_ * 2), (bf16_t*)alloc((size_t)B_ * H_ * 2) };
  bf16_t* h2[2] = { (bf16_t*)alloc((size_t)B_ * H_ * 2), (bf16_t*)alloc((size_t)B_ * H_ * 2) };
  bf16_t* h3[2] = { (bf16_t*)alloc((size_t)B_ * H_ * 2), (bf16_t*)alloc((size_t)B_ * H_ * 2) };
  float*  c1    = (float*)alloc((size_t)B_ * H_ * 4);
  float*  c2    = (float*)alloc((size_t)B_ * H_ * 4);
  float*  c3    = (float*)alloc((size_t)B_ * H_ * 4);
  unsigned* bar = (unsigned*)alloc(256);

  // --- prep: convert inputs/weights to bf16, fuse biases, zero state ---
  k_convert_x  <<<2048, 256, 0, stream>>>(x, xp);
  k_convert_pad<<<512,  256, 0, stream>>>(Wih1, Wih1b, G4_, IN_, INP_);
  k_convert_pad<<<1024, 256, 0, stream>>>(Whh1, Whh1b, G4_, H_, H_);
  k_convert_pad<<<1024, 256, 0, stream>>>(Wih2, Wih2b, G4_, H_, H_);
  k_convert_pad<<<1024, 256, 0, stream>>>(Whh2, Whh2b, G4_, H_, H_);
  k_convert_pad<<<1024, 256, 0, stream>>>(Wih3, Wih3b, G4_, H_, H_);
  k_convert_pad<<<1024, 256, 0, stream>>>(Whh3, Whh3b, G4_, H_, H_);
  k_convert_pad<<<64,   256, 0, stream>>>(Wo,   Wob,   OUT_, H_, H_);
  k_bias_add<<<16, 256, 0, stream>>>(bih1, bhh1, b1, G4_);
  k_bias_add<<<16, 256, 0, stream>>>(bih2, bhh2, b2, G4_);
  k_bias_add<<<16, 256, 0, stream>>>(bih3, bhh3, b3, G4_);
  k_bias_add<<<1, 128, 0, stream>>>(bo, nullptr, bob, OUT_);
  k_zero_f32 <<<64, 256, 0, stream>>>(c1, (size_t)B_ * H_);
  k_zero_f32 <<<64, 256, 0, stream>>>(c2, (size_t)B_ * H_);
  k_zero_f32 <<<64, 256, 0, stream>>>(c3, (size_t)B_ * H_);
  k_zero_bf16<<<64, 256, 0, stream>>>(h1[0], (size_t)B_ * H_);
  k_zero_bf16<<<64, 256, 0, stream>>>(h2[0], (size_t)B_ * H_);
  k_zero_bf16<<<64, 256, 0, stream>>>(h3[0], (size_t)B_ * H_);
  k_out_t0   <<<32, 256, 0, stream>>>(out);
  k_zero_u32 <<<1, 32, 0, stream>>>(bar, 8);

  // --- persistent LSTM scan ---
  KParams P;
  P.xp = xp;
  P.Wih1 = Wih1b; P.Whh1 = Whh1b;
  P.Wih2 = Wih2b; P.Whh2 = Whh2b;
  P.Wih3 = Wih3b; P.Whh3 = Whh3b;
  P.Wo = Wob;
  P.b1 = b1; P.b2 = b2; P.b3 = b3; P.bo = bob;
  P.h1[0] = h1[0]; P.h1[1] = h1[1];
  P.h2[0] = h2[0]; P.h2[1] = h2[1];
  P.h3[0] = h3[0]; P.h3[1] = h3[1];
  P.c1 = c1; P.c2 = c2; P.c3 = c3;
  P.out = out;
  P.bar = bar;

  lstm_persistent<<<NWG, NTHR, 0, stream>>>(P);
}
